// MoE_61615600829044
// MI455X (gfx1250) — matrix operand order repaired and checked by reading, not compiled, not run on hardware
//
#include <hip/hip_runtime.h>

// ---------------- problem constants ----------------
#define T_TOK 16384   // B*S
#define DDIM  1024
#define FDIM  4096
#define NEXP  8
#define MT    32      // token tile rows
#define MAXT  (T_TOK / MT)   // 512 tiles per expert worst case
#define FC    128     // F chunk

typedef __attribute__((ext_vector_type(16))) __bf16 v16bf;
typedef __attribute__((ext_vector_type(8)))  __bf16 v8bf;
typedef __attribute__((ext_vector_type(4)))  __bf16 v4bf;
typedef __attribute__((ext_vector_type(2)))  __bf16 v2bf;
typedef __attribute__((ext_vector_type(8)))  float  v8f;

__device__ __forceinline__ float gelu_exact(float v) {
    return 0.5f * v * (1.0f + erff(v * 0.70710678118654752440f));
}

__device__ __forceinline__ v16bf cat16(v8bf lo, v8bf hi) {
    v16bf r;
#pragma unroll
    for (int i = 0; i < 8; ++i) { r[i] = lo[i]; r[i + 8] = hi[i]; }
    return r;
}

// ---------------- prep: x fp32 -> bf16 ----------------
__global__ __launch_bounds__(256) void cvt_x_bf16(const float* __restrict__ in,
                                                  __bf16* __restrict__ out)
{
    size_t idx = (size_t)blockIdx.x * 256 + threadIdx.x;   // one float4 each
    float4 v = ((const float4*)in)[idx];
    v4bf o = {(__bf16)v.x, (__bf16)v.y, (__bf16)v.z, (__bf16)v.w};
    ((v4bf*)out)[idx] = o;
}

// -------- prep: transpose + convert  in[R][C] fp32 -> outT[C][R] bf16 (per expert z) --------
__global__ __launch_bounds__(256) void transpose_cvt_bf16(const float* __restrict__ in,
                                                          __bf16* __restrict__ outT,
                                                          int R, int C)
{
    __shared__ float tile[64][65];
    const int r0 = blockIdx.y * 64, c0 = blockIdx.x * 64;
    const float* ip = in   + (size_t)blockIdx.z * R * C;
    __bf16*      op = outT + (size_t)blockIdx.z * R * C;
#pragma unroll
    for (int i = 0; i < 16; ++i) {
        int lin = i * 256 + threadIdx.x;        // 0..4095
        int r = lin >> 6, c = lin & 63;
        tile[r][c] = ip[(size_t)(r0 + r) * C + c0 + c];
    }
    __syncthreads();
#pragma unroll
    for (int i = 0; i < 8; ++i) {               // write pairs along R (contiguous in outT)
        int lin = i * 256 + threadIdx.x;        // 0..2047
        int c = lin >> 5, rp = (lin & 31) * 2;
        v2bf o = {(__bf16)tile[rp][c], (__bf16)tile[rp + 1][c]};
        *(v2bf*)&op[(size_t)(c0 + c) * R + r0 + rp] = o;
    }
}

// ---------------- kernel 1: router (wave per token) ----------------
__global__ __launch_bounds__(256) void moe_router(
    const float* __restrict__ x, const float* __restrict__ rw,
    const float* __restrict__ rb, int* __restrict__ topi, float* __restrict__ topv)
{
    const int lane  = threadIdx.x & 31;
    const int token = blockIdx.x * 8 + (threadIdx.x >> 5);
    if (token >= T_TOK) return;

    const int eL = lane & 7;
    const int dL = lane >> 3;
    const float* xr = x + (size_t)token * DDIM;

    float acc = 0.0f;
    for (int i = 0; i < DDIM / 4; ++i) {
        int d = dL + 4 * i;
        acc = fmaf(xr[d], rw[d * NEXP + eL], acc);
    }
    acc += __shfl_xor(acc, 8);
    acc += __shfl_xor(acc, 16);
    float logit = acc + rb[eL];

    float mx = logit;
    mx = fmaxf(mx, __shfl_xor(mx, 1));
    mx = fmaxf(mx, __shfl_xor(mx, 2));
    mx = fmaxf(mx, __shfl_xor(mx, 4));
    float p = __expf(logit - mx);
    float s = p;
    s += __shfl_xor(s, 1); s += __shfl_xor(s, 2); s += __shfl_xor(s, 4);
    p /= s;

    float v1 = p; int i1 = eL;
#pragma unroll
    for (int o = 1; o <= 4; o <<= 1) {
        float ov = __shfl_xor(v1, o); int oi = __shfl_xor(i1, o);
        if (ov > v1 || (ov == v1 && oi < i1)) { v1 = ov; i1 = oi; }
    }
    float pm = (eL == i1) ? -1.0f : p;
    float v2 = pm; int i2 = eL;
#pragma unroll
    for (int o = 1; o <= 4; o <<= 1) {
        float ov = __shfl_xor(v2, o); int oi = __shfl_xor(i2, o);
        if (ov > v2 || (ov == v2 && oi < i2)) { v2 = ov; i2 = oi; }
    }
    if (lane == 0) {
        topi[token * 2]     = i1; topv[token * 2]     = v1;
        topi[token * 2 + 1] = i2; topv[token * 2 + 1] = v2;
    }
}

// ------------- kernel 2: stable per-expert compaction -------------
__global__ __launch_bounds__(256) void moe_compact(
    const int* __restrict__ topi, const float* __restrict__ topv,
    int* __restrict__ pairTok, float* __restrict__ pairGate, int* __restrict__ counts)
{
    const int ex   = threadIdx.x >> 5;
    const int lane = threadIdx.x & 31;
    int cnt = 0;
    for (int base = 0; base < T_TOK; base += 32) {
        int tok = base + lane;
        int a = topi[tok * 2], b = topi[tok * 2 + 1];
        bool sel = (a == ex) || (b == ex);
        float g  = (a == ex) ? topv[tok * 2] : topv[tok * 2 + 1];
        unsigned mask = (unsigned)__ballot(sel);
        if (sel) {
            int off = __popc(mask & ((1u << lane) - 1u));
            pairTok [ex * T_TOK + cnt + off] = tok;
            pairGate[ex * T_TOK + cnt + off] = g;
        }
        cnt += __popc(mask);
    }
    if (lane == 0) counts[ex] = cnt;
}

// ------------- kernel 3 (fast path): fused FFN on pre-converted bf16 operands -------------
// xb:  [T][D]    bf16 ; w1t: [E][F][D] bf16 ; w2t: [E][D][F] bf16
__global__ __launch_bounds__(512) void moe_ffn_bf16(
    const __bf16* __restrict__ xb,  const __bf16* __restrict__ w1t,
    const float* __restrict__ b1,   const __bf16* __restrict__ w2t,
    const float* __restrict__ b2,   const int* __restrict__ pairTok,
    const float* __restrict__ pairGate, const int* __restrict__ counts,
    float* __restrict__ out)
{
    __shared__ __align__(32) __bf16 sH[MT * 144];
    __shared__ int   sTok [MT];
    __shared__ float sGate[MT];

    const int e    = blockIdx.x >> 9;
    const int tile = blockIdx.x & (MAXT - 1);
    const int cnt  = counts[e];
    const int row0 = tile * MT;
    if (row0 >= cnt) return;

    const int tid  = threadIdx.x;
    const int lane = tid & 31;
    const int wid  = tid >> 5;
    const int hi   = lane >> 4;
    const int l15  = lane & 15;

    if (tid < MT) {
        int r = row0 + tid;
        bool ok = (r < cnt);
        sTok [tid] = pairTok [e * T_TOK + (ok ? r : row0)];
        sGate[tid] = ok ? pairGate[e * T_TOK + r] : 0.0f;
    }
    __syncthreads();

    const v8f vzero = {0.f,0.f,0.f,0.f,0.f,0.f,0.f,0.f};
    v8f yacc[2][4];
#pragma unroll
    for (int i = 0; i < 2; ++i)
#pragma unroll
        for (int j = 0; j < 4; ++j) yacc[i][j] = vzero;

    const int mt1 = wid >> 3;                  // GEMM1 row tile
    const int nq  = wid & 7;                   // GEMM1 16-col slot

    // per-lane invariant A row pointer (contiguous 16-bit A layout: K = {hi*16..+15})
    const int atok = sTok[mt1 * 16 + l15];
    const __bf16* ap  = xb + (size_t)atok * DDIM + hi * 16;
    const __bf16* wp2 = w2t + ((size_t)e * DDIM + wid * 64 + l15) * FDIM + hi * 16;

    for (int fc = 0; fc < FDIM; fc += FC) {
        // ---- GEMM1: H[32][FC] = X @ W1[:, fc:fc+FC]   (no LDS, software pipelined)
        const __bf16* bp = w1t + ((size_t)e * FDIM + fc + nq * 16 + l15) * DDIM + hi * 16;
        if (fc + FC < FDIM) {
            __builtin_prefetch(bp + (size_t)FC * DDIM, 0, 3);          // next W1 cols
            __builtin_prefetch(wp2 + fc + FC, 0, 3);                   // next W2 rows
        }
        v8f hacc = vzero;
        v16bf af = cat16(*(const v8bf*)(ap), *(const v8bf*)(ap + 8));
        v16bf bf = *(const v16bf*)(bp);
#pragma unroll 8
        for (int ks = 0; ks < DDIM / 32; ++ks) {
            v16bf afn, bfn;
            if (ks < DDIM / 32 - 1) {
                const int d1 = (ks + 1) * 32;
                afn = cat16(*(const v8bf*)(ap + d1), *(const v8bf*)(ap + d1 + 8));
                bfn = *(const v16bf*)(bp + d1);
            }
            hacc = __builtin_amdgcn_wmma_f32_16x16x32_bf16(
                false, af, false, bf, (short)0, hacc, false, false);
            af = afn; bf = bfn;
        }
        // ---- bias + GELU -> sH (bf16)
#pragma unroll
        for (int r = 0; r < 8; ++r) {
            int m = mt1 * 16 + r + 8 * hi;
            int n = nq * 16 + l15;
            float v = hacc[r] + b1[e * FDIM + fc + n];
            sH[m * 144 + n] = (__bf16)gelu_exact(v);
        }
        __syncthreads();
        // ---- GEMM2 partial: Y[32][wid*64 .. +64] += H @ W2[fc:fc+FC, :]
        // double-buffered across ks2: loads for ks2+1 in flight while ks2's WMMAs run
        v16bf bfr[4], a2[2];
#pragma unroll
        for (int nt = 0; nt < 4; ++nt)
            bfr[nt] = *(const v16bf*)(wp2 + (size_t)(nt * 16) * FDIM + fc);
#pragma unroll
        for (int mt2 = 0; mt2 < 2; ++mt2) {
            int m = mt2 * 16 + l15;
            a2[mt2] = cat16(*(const v8bf*)&sH[m * 144 + hi * 16],
                            *(const v8bf*)&sH[m * 144 + hi * 16 + 8]);
        }
#pragma unroll
        for (int ks2 = 0; ks2 < 4; ++ks2) {
            v16bf bnx[4], anx[2];
            if (ks2 < 3) {
                const int kb = (ks2 + 1) * 32;
#pragma unroll
                for (int nt = 0; nt < 4; ++nt)
                    bnx[nt] = *(const v16bf*)(wp2 + (size_t)(nt * 16) * FDIM + fc + kb);
#pragma unroll
                for (int mt2 = 0; mt2 < 2; ++mt2) {
                    int m = mt2 * 16 + l15;
                    anx[mt2] = cat16(*(const v8bf*)&sH[m * 144 + kb + hi * 16],
                                     *(const v8bf*)&sH[m * 144 + kb + hi * 16 + 8]);
                }
            }
#pragma unroll
            for (int nt = 0; nt < 4; ++nt) {
                yacc[0][nt] = __builtin_amdgcn_wmma_f32_16x16x32_bf16(
                    false, a2[0], false, bfr[nt], (short)0, yacc[0][nt], false, false);
                yacc[1][nt] = __builtin_amdgcn_wmma_f32_16x16x32_bf16(
                    false, a2[1], false, bfr[nt], (short)0, yacc[1][nt], false, false);
            }
#pragma unroll
            for (int nt = 0; nt < 4; ++nt) bfr[nt] = bnx[nt];
            a2[0] = anx[0]; a2[1] = anx[1];
        }
        __syncthreads();
    }
    // ---- epilogue: out[tok][n] += gate * (y + b2)  (2 float adds/element, commutative)
#pragma unroll
    for (int mt2 = 0; mt2 < 2; ++mt2) {
#pragma unroll
        for (int nt = 0; nt < 4; ++nt) {
#pragma unroll
            for (int r = 0; r < 8; ++r) {
                int m = mt2 * 16 + r + 8 * hi;
                int n = wid * 64 + nt * 16 + l15;
                float g = sGate[m];
                if (g != 0.0f) {
                    float v = yacc[mt2][nt][r] + b2[e * DDIM + n];
                    atomicAdd(&out[(size_t)sTok[m] * DDIM + n], g * v);
                }
            }
        }
    }
}

// ------------- kernel 3 (fallback, small workspace): fp32 operands, per-lane gather -------------
__global__ __launch_bounds__(512) void moe_ffn_fp32(
    const float* __restrict__ x,  const float* __restrict__ w1,
    const float* __restrict__ b1, const float* __restrict__ w2,
    const float* __restrict__ b2, const int* __restrict__ pairTok,
    const float* __restrict__ pairGate, const int* __restrict__ counts,
    float* __restrict__ out)
{
    __shared__ __align__(32) __bf16 sX [MT * 48];
    __shared__ __align__(32) __bf16 sB1[FC * 48];
    __shared__ __align__(32) __bf16 sH [MT * 144];
    __shared__ int   sTok [MT];
    __shared__ float sGate[MT];

    const int e    = blockIdx.x >> 9;
    const int tile = blockIdx.x & (MAXT - 1);
    const int cnt  = counts[e];
    const int row0 = tile * MT;
    if (row0 >= cnt) return;

    const int tid  = threadIdx.x;
    const int lane = tid & 31;
    const int wid  = tid >> 5;
    const int hi   = lane >> 4;
    const int l15  = lane & 15;

    if (tid < MT) {
        int r = row0 + tid;
        bool ok = (r < cnt);
        sTok [tid] = pairTok [e * T_TOK + (ok ? r : row0)];
        sGate[tid] = ok ? pairGate[e * T_TOK + r] : 0.0f;
    }
    __syncthreads();

    const v8f vzero = {0.f,0.f,0.f,0.f,0.f,0.f,0.f,0.f};
    v8f yacc[2][4];
#pragma unroll
    for (int i = 0; i < 2; ++i)
#pragma unroll
        for (int j = 0; j < 4; ++j) yacc[i][j] = vzero;

    const int mt1 = wid >> 3;
    const int nq  = wid & 7;
    const size_t w1base = (size_t)e * DDIM * FDIM;
    const size_t w2base = (size_t)e * FDIM * DDIM;

    for (int fc = 0; fc < FDIM; fc += FC) {
        v8f hacc = vzero;
        for (int ks = 0; ks < DDIM / 32; ++ks) {
            const int d0 = ks * 32;
            {
                int m  = tid >> 4;
                int kk = (tid & 15) * 2;
                const float* xp = x + (size_t)sTok[m] * DDIM + d0 + kk;
                sX[m * 48 + kk]     = (__bf16)xp[0];
                sX[m * 48 + kk + 1] = (__bf16)xp[1];
            }
#pragma unroll
            for (int i = 0; i < 8; ++i) {
                int linear = i * 512 + tid;
                int k = linear >> 7;
                int n = linear & 127;
                sB1[n * 48 + k] =
                    (__bf16)w1[w1base + (size_t)(d0 + k) * FDIM + fc + n];
            }
            __syncthreads();
            int am = mt1 * 16 + l15;
            v16bf afrag = cat16(*(const v8bf*)&sX[am * 48 + hi * 16],
                                *(const v8bf*)&sX[am * 48 + hi * 16 + 8]);
            v16bf bfrag = *(const v16bf*)&sB1[(nq * 16 + l15) * 48 + hi * 16];
            hacc = __builtin_amdgcn_wmma_f32_16x16x32_bf16(
                false, afrag, false, bfrag, (short)0, hacc, false, false);
            __syncthreads();
        }
#pragma unroll
        for (int r = 0; r < 8; ++r) {
            int m = mt1 * 16 + r + 8 * hi;
            int n = nq * 16 + l15;
            float v = hacc[r] + b1[e * FDIM + fc + n];
            sH[m * 144 + n] = (__bf16)gelu_exact(v);
        }
        __syncthreads();
#pragma unroll
        for (int ks2 = 0; ks2 < 4; ++ks2) {
            v16bf a2[2];
#pragma unroll
            for (int mt2 = 0; mt2 < 2; ++mt2) {
                int m = mt2 * 16 + l15;
                a2[mt2] = cat16(*(const v8bf*)&sH[m * 144 + ks2 * 32 + hi * 16],
                                *(const v8bf*)&sH[m * 144 + ks2 * 32 + hi * 16 + 8]);
            }
#pragma unroll
            for (int nt = 0; nt < 4; ++nt) {
                int n = wid * 64 + nt * 16 + l15;
                const float* wp = w2 + w2base +
                                  (size_t)(fc + ks2 * 32 + hi * 16) * DDIM + n;
                v16bf bfrag;
#pragma unroll
                for (int i = 0; i < 16; ++i) bfrag[i] = (__bf16)wp[(size_t)i * DDIM];
                yacc[0][nt] = __builtin_amdgcn_wmma_f32_16x16x32_bf16(
                    false, a2[0], false, bfrag, (short)0, yacc[0][nt], false, false);
                yacc[1][nt] = __builtin_amdgcn_wmma_f32_16x16x32_bf16(
                    false, a2[1], false, bfrag, (short)0, yacc[1][nt], false, false);
            }
        }
        __syncthreads();
    }
#pragma unroll
    for (int mt2 = 0; mt2 < 2; ++mt2) {
#pragma unroll
        for (int nt = 0; nt < 4; ++nt) {
#pragma unroll
            for (int r = 0; r < 8; ++r) {
                int m = mt2 * 16 + r + 8 * hi;
                int n = wid * 64 + nt * 16 + l15;
                float g = sGate[m];
                if (g != 0.0f) {
                    float v = yacc[mt2][nt][r] + b2[e * DDIM + n];
                    atomicAdd(&out[(size_t)sTok[m] * DDIM + n], g * v);
                }
            }
        }
    }
}

// ---------------- launcher ----------------
extern "C" void kernel_launch(void* const* d_in, const int* in_sizes, int n_in,
                              void* d_out, int out_size, void* d_ws, size_t ws_size,
                              hipStream_t stream) {
    (void)in_sizes; (void)n_in;
    const float* x  = (const float*)d_in[0];
    const float* rw = (const float*)d_in[1];
    const float* rb = (const float*)d_in[2];
    const float* w1 = (const float*)d_in[3];
    const float* b1 = (const float*)d_in[4];
    const float* w2 = (const float*)d_in[5];
    const float* b2 = (const float*)d_in[6];
    float* out = (float*)d_out;

    char* ws = (char*)d_ws;
    // routing metadata (~1.25 MB)
    int*   topi     = (int*)  (ws + 0);
    float* topv     = (float*)(ws + 131072);
    int*   pairTok  = (int*)  (ws + 262144);
    float* pairGate = (float*)(ws + 786432);
    int*   counts   = (int*)  (ws + 1310720);
    // bf16 operand buffers
    const size_t OFF_XB  = 1310976;                              // 256-aligned
    const size_t SZ_XB   = (size_t)T_TOK * DDIM * 2;             // 32 MB
    const size_t OFF_W1T = OFF_XB + SZ_XB;
    const size_t SZ_W1T  = (size_t)NEXP * DDIM * FDIM * 2;       // 64 MB
    const size_t OFF_W2T = OFF_W1T + SZ_W1T;
    const size_t SZ_W2T  = (size_t)NEXP * FDIM * DDIM * 2;       // 64 MB
    const size_t NEED    = OFF_W2T + SZ_W2T;

    hipMemsetAsync(d_out, 0, (size_t)out_size * sizeof(float), stream);

    moe_router <<<T_TOK / 8, 256, 0, stream>>>(x, rw, rb, topi, topv);
    moe_compact<<<1, 256, 0, stream>>>(topi, topv, pairTok, pairGate, counts);

    if (ws_size >= NEED) {
        __bf16* xb  = (__bf16*)(ws + OFF_XB);
        __bf16* w1t = (__bf16*)(ws + OFF_W1T);
        __bf16* w2t = (__bf16*)(ws + OFF_W2T);
        cvt_x_bf16<<<(T_TOK * DDIM) / 4 / 256, 256, 0, stream>>>(x, xb);
        // w1 [E][D][F] -> w1t [E][F][D]
        transpose_cvt_bf16<<<dim3(FDIM / 64, DDIM / 64, NEXP), 256, 0, stream>>>(
            w1, w1t, DDIM, FDIM);
        // w2 [E][F][D] -> w2t [E][D][F]
        transpose_cvt_bf16<<<dim3(DDIM / 64, FDIM / 64, NEXP), 256, 0, stream>>>(
            w2, w2t, FDIM, DDIM);
        moe_ffn_bf16<<<NEXP * MAXT, 512, 0, stream>>>(xb, w1t, b1, w2t, b2,
                                                      pairTok, pairGate, counts, out);
    } else {
        moe_ffn_fp32<<<NEXP * MAXT, 512, 0, stream>>>(x, w1, b1, w2, b2,
                                                      pairTok, pairGate, counts, out);
    }
}
